// GraphCare_89936615178714
// MI455X (gfx1250) — matrix operand order repaired and checked by reading, not compiled, not run on hardware
//
#include <hip/hip_runtime.h>

#define BN   16
#define VVIS 30
#define NVV  1501
#define NVP  1504
#define NEE  2001
#define HH   128
#define LLY  2
#define OUTD 25
#define NTOT 100000
#define ETOT 800000
#define NPB  (NTOT / BN)

typedef __bf16 bf16_t;
typedef __attribute__((ext_vector_type(16))) __bf16 v16bf;
typedef __attribute__((ext_vector_type(8)))  float  v8f;

// ---------------- fragment loaders (CDNA5 §7.12.2 layouts) -----------------
// Rows are guaranteed in-bounds by operand padding -> unconditional loads,
// no EXEC-mask branching in the WMMA k-loop.
// A-matrix 16x32 bf16: lanes 0-15 -> rows, K = {khalf+0..7, 16+khalf+0..7}
__device__ __forceinline__ v16bf load_fragA(const bf16_t* __restrict__ base, int ld,
                                            int row, int kc, int lane) {
  int khalf = (lane & 16) ? 8 : 0;
  const bf16_t* p = base + (size_t)row * ld + kc + khalf;
  v16bf f;
#pragma unroll
  for (int j = 0; j < 8; ++j) { f[j] = p[j]; f[j + 8] = p[16 + j]; }
  return f;
}
// B-matrix 32x16 bf16 (row-major [N][K], column n = row n):
// lanes 0-15 hold K={0..7,16..23}, lanes 16-31 hold K={8..15,24..31} for column n = lane&15
__device__ __forceinline__ v16bf load_fragB(const bf16_t* __restrict__ base, int ld,
                                            int row, int kc, int lane) {
  const bf16_t* p = base + (size_t)row * ld + kc + ((lane & 16) ? 8 : 0);
  v16bf f;
#pragma unroll
  for (int j = 0; j < 8; ++j) { f[j] = p[j]; f[j + 8] = p[16 + j]; }
  return f;
}

// ---------------- generic C = A @ B^T + bias (bf16 in, f32 out, WMMA) ------
// A has >= ceil16(M) zero-padded rows; Bm has >= ceil16(N) zero-padded rows.
__global__ void gemm_abt_kernel(const bf16_t* __restrict__ A, int lda, int M,
                                const bf16_t* __restrict__ Bm, int ldb, int N,
                                int K, const float* __restrict__ bias,
                                float* __restrict__ C, int ldc) {
  int lane  = threadIdx.x & 31;
  int wave  = threadIdx.x >> 5;
  int mTile = blockIdx.x * (blockDim.x >> 5) + wave;
  int mBase = mTile * 16;
  int nBase = blockIdx.y * 16;
  if (mBase >= ((M + 15) & ~15)) return;  // wave-uniform
  v8f acc = {};
  for (int kc = 0; kc < K; kc += 32) {
    v16bf a = load_fragA(A,  lda, mBase + (lane & 15), kc, lane);
    v16bf b = load_fragB(Bm, ldb, nBase + (lane & 15), kc, lane);
    acc = __builtin_amdgcn_wmma_f32_16x16x32_bf16(false, a, false, b, (short)0, acc, false, false);
  }
  int n = nBase + (lane & 15);
  if (n < N) {
    float bn = bias ? bias[n] : 0.0f;
    int mOff = (lane & 16) ? 8 : 0;
#pragma unroll
    for (int r = 0; r < 8; ++r) {
      int m = mBase + r + mOff;
      if (m < M) C[(size_t)m * ldc + n] = acc[r] + bn;
    }
  }
}

// ---------------- conv layer: x = relu((agg + x) @ W^T + b), in-place ------
__global__ void conv_kernel(float* __restrict__ x, const float* __restrict__ agg,
                            const bf16_t* __restrict__ Wbf, const float* __restrict__ bias) {
  int lane  = threadIdx.x & 31;
  int wave  = threadIdx.x >> 5;
  int mTile = blockIdx.x * 8 + wave;
  int mBase = mTile * 16;
  if (mBase >= NTOT) return;
  int row   = mBase + (lane & 15);
  int khalf = (lane & 16) ? 8 : 0;
  const float* xr = x   + (size_t)row * HH;
  const float* ar = agg + (size_t)row * HH;
  v16bf afr[4];
#pragma unroll
  for (int c = 0; c < 4; ++c) {
    int k0 = c * 32 + khalf;
#pragma unroll
    for (int j = 0; j < 8; ++j) {
      afr[c][j]     = (bf16_t)(xr[k0 + j]      + ar[k0 + j]);
      afr[c][j + 8] = (bf16_t)(xr[k0 + 16 + j] + ar[k0 + 16 + j]);
    }
  }
#pragma unroll
  for (int nt = 0; nt < 8; ++nt) {
    v8f acc = {};
#pragma unroll
    for (int c = 0; c < 4; ++c) {
      v16bf b = load_fragB(Wbf, HH, nt * 16 + (lane & 15), c * 32, lane);
      acc = __builtin_amdgcn_wmma_f32_16x16x32_bf16(false, afr[c], false, b, (short)0, acc, false, false);
    }
    int n = nt * 16 + (lane & 15);
    float bn = bias[n];
    int mOff = (lane & 16) ? 8 : 0;
#pragma unroll
    for (int r = 0; r < 8; ++r) {
      int m = mBase + r + mOff;
      x[(size_t)m * HH + n] = fmaxf(acc[r] + bn, 0.0f);
    }
  }
}

// ---------------- small conversion / table kernels -------------------------
__global__ void cvt_bf16_kernel(const float* __restrict__ in, bf16_t* __restrict__ out, long n) {
  long i = (long)blockIdx.x * blockDim.x + threadIdx.x;
  if (i < n) out[i] = (bf16_t)in[i];
}
// node_emb padded to NVP rows (zero rows >= NVV)
__global__ void cvt_node_kernel(const float* __restrict__ in, bf16_t* __restrict__ out) {
  long i = (long)blockIdx.x * blockDim.x + threadIdx.x;
  if (i >= (long)NVP * HH) return;
  long n = i / HH; int k = (int)(i % HH);
  out[i] = (bf16_t)((n < NVV) ? in[n * HH + k] : 0.0f);
}
__global__ void cvt_vn_kernel(const int* __restrict__ vn, bf16_t* __restrict__ out) {
  long i = (long)blockIdx.x * blockDim.x + threadIdx.x;
  if (i >= (long)BN * VVIS * NVP) return;
  long r = i / NVP; int k = (int)(i % NVP);
  out[i] = (bf16_t)((k < NVV) ? (float)vn[r * NVV + k] : 0.0f);
}
// alpha padded to NVP rows x NVP cols per layer
__global__ void cvt_alpha_kernel(const float* __restrict__ aw, bf16_t* __restrict__ out) {
  long i = (long)blockIdx.x * blockDim.x + threadIdx.x;
  if (i >= (long)LLY * NVP * NVP) return;
  long l = i / ((long)NVP * NVP);
  long rem = i % ((long)NVP * NVP);
  long n = rem / NVP; int k = (int)(rem % NVP);
  out[i] = (bf16_t)((n < NVV && k < NVV) ? aw[(l * NVV + n) * NVV + k] : 0.0f);
}
// u[l][d] = sum_h wr_w[l][h]*lin_w[h][d];  c[l] = lin_b . wr_w[l] + wr_b[l]
__global__ void u_kernel(const float* __restrict__ lin_w, const float* __restrict__ lin_b,
                         const float* __restrict__ wr_w, const float* __restrict__ wr_b,
                         float* __restrict__ u, float* __restrict__ c) {
  int l = blockIdx.x, d = threadIdx.x;
  float s = 0.0f;
  for (int h = 0; h < HH; ++h) s += wr_w[l * HH + h] * lin_w[h * HH + d];
  u[l * HH + d] = s;
  if (d == 0) {
    float cc = wr_b[l];
    for (int h = 0; h < HH; ++h) cc += lin_b[h] * wr_w[l * HH + h];
    c[l] = cc;
  }
}
__global__ void wrel_kernel(const float* __restrict__ edge_emb, const float* __restrict__ u,
                            const float* __restrict__ c, float* __restrict__ wrel) {
  int i = blockIdx.x * blockDim.x + threadIdx.x;
  if (i >= LLY * NEE) return;
  int l = i / NEE, e = i % NEE;
  const float* er = edge_emb + (size_t)e * HH;
  const float* ur = u + l * HH;
  float s = c[l];
  for (int d = 0; d < HH; ++d) s += er[d] * ur[d];
  wrel[i] = 1.0f / (1.0f + __expf(-s));
}
__global__ void beta_kernel(const int* __restrict__ vn, const float* __restrict__ beta_w,
                            const float* __restrict__ beta_b, float* __restrict__ beta) {
  int i = blockIdx.x * blockDim.x + threadIdx.x;
  if (i >= LLY * BN * VVIS) return;
  int l = i / (BN * VVIS);
  int rem = i % (BN * VVIS);
  int b = rem / VVIS, v = rem % VVIS;
  const int* row = vn + ((size_t)b * VVIS + v) * NVV;
  const float* w = beta_w + l * NVV;
  float s = beta_b[l];
  for (int n = 0; n < NVV; ++n) s += (float)row[n] * w[n];
  beta[i] = tanhf(s) * __expf(0.01f * (float)(VVIS - v));
}
// attn[b][n] = sum_v softmax_v(logits[b,v,n]) * beta[b][v]
__global__ void attn_kernel(const float* __restrict__ logits, const float* __restrict__ beta_l,
                            float* __restrict__ attn_l) {
  int i = blockIdx.x * blockDim.x + threadIdx.x;
  if (i >= BN * NVV) return;
  int b = i / NVV, n = i % NVV;
  const float* lp = logits + (size_t)b * VVIS * NVP + n;
  float mx = -1e30f;
  for (int v = 0; v < VVIS; ++v) mx = fmaxf(mx, lp[(size_t)v * NVP]);
  float se = 0.0f, acc = 0.0f;
  for (int v = 0; v < VVIS; ++v) {
    float ev = __expf(lp[(size_t)v * NVP] - mx);
    se += ev;
    acc += ev * beta_l[b * VVIS + v];
  }
  attn_l[i] = acc / se;
}
__global__ void xinit_kernel(const float4* __restrict__ xproj, const int* __restrict__ nid,
                             float4* __restrict__ x) {
  long idx = (long)blockIdx.x * blockDim.x + threadIdx.x;
  if (idx >= (long)NTOT * 32) return;
  long i = idx >> 5; int q = (int)(idx & 31);
  x[idx] = xproj[(size_t)nid[i] * 32 + q];
}

// ---------------- edge message pass + scatter-add --------------------------
__global__ void message_kernel(const float* __restrict__ x, float* __restrict__ agg,
                               const int* __restrict__ src, const int* __restrict__ dst,
                               const int* __restrict__ nid, const int* __restrict__ eid,
                               const int* __restrict__ bat,
                               const float* __restrict__ attn_l, const float* __restrict__ wrel_l) {
  __shared__ float s_coef[128];
  __shared__ int   s_src[128];
  __shared__ int   s_dst[128];
  int lane = threadIdx.x & 31;
  int wave = threadIdx.x >> 5;
  for (long base = (long)blockIdx.x * 128; base < ETOT; base += (long)gridDim.x * 128) {
    long e = base + threadIdx.x;
    if (e < ETOT) {
      int s = src[e];
      s_src[threadIdx.x]  = s;
      s_dst[threadIdx.x]  = dst[e];
      s_coef[threadIdx.x] = attn_l[(size_t)bat[s] * NVV + nid[s]] * wrel_l[eid[e]];
    }
    __syncthreads();
    int cnt = (int)((ETOT - base) < 128 ? (ETOT - base) : 128);
    for (int j = wave; j < cnt; j += 4) {
      float cf = s_coef[j];
      float4 v = ((const float4*)(x + (size_t)s_src[j] * HH))[lane];
      float* ab = agg + (size_t)s_dst[j] * HH + lane * 4;
      atomicAdd(ab + 0, fmaxf(v.x * cf, 0.0f));
      atomicAdd(ab + 1, fmaxf(v.y * cf, 0.0f));
      atomicAdd(ab + 2, fmaxf(v.z * cf, 0.0f));
      atomicAdd(ab + 3, fmaxf(v.w * cf, 0.0f));
    }
    __syncthreads();
  }
}

// ---------------- pooling / node branch / head -----------------------------
__global__ void pool_kernel(const float* __restrict__ x, float* __restrict__ xgraph) {
  int b = blockIdx.x, h = threadIdx.x;
  float s = 0.0f;
  for (int i = blockIdx.y; i < NPB; i += 8)
    s += x[((size_t)b * NPB + i) * HH + h];
  atomicAdd(&xgraph[b * HH + h], s * (1.0f / (float)NPB));
}
__global__ void xnode_emb_kernel(const int* __restrict__ ehr, const float* __restrict__ node_emb,
                                 float* __restrict__ xne) {
  int b = blockIdx.x, d = threadIdx.x;
  float acc = 0.0f, cnt = 0.0f;
  for (int n = 0; n < NVV; ++n) {
    int e = ehr[b * NVV + n];
    if (e) { cnt += (float)e; acc += (float)e * node_emb[(size_t)n * HH + d]; }
  }
  xne[b * HH + d] = acc / cnt;
}
__global__ void xnode_proj_kernel(const float* __restrict__ xne, const float* __restrict__ lin_w,
                                  const float* __restrict__ lin_b, float* __restrict__ xnode) {
  int i = blockIdx.x * blockDim.x + threadIdx.x;
  if (i >= BN * HH) return;
  int b = i >> 7, h = i & 127;
  float s = lin_b[h];
  for (int d = 0; d < HH; ++d) s += xne[b * HH + d] * lin_w[h * HH + d];
  xnode[i] = s;
}
__global__ void final_kernel(const float* __restrict__ xgraph, const float* __restrict__ xnode,
                             const float* __restrict__ mlp_w, const float* __restrict__ mlp_b,
                             float* __restrict__ out) {
  int i = blockIdx.x * blockDim.x + threadIdx.x;
  if (i >= BN * OUTD) return;
  int b = i / OUTD, o = i % OUTD;
  float s = mlp_b[o];
  for (int h = 0; h < HH; ++h) s += xgraph[b * HH + h] * mlp_w[o * 2 * HH + h];
  for (int h = 0; h < HH; ++h) s += xnode[b * HH + h] * mlp_w[o * 2 * HH + HH + h];
  out[i] = s;
}

// ---------------------------------------------------------------------------
extern "C" void kernel_launch(void* const* d_in, const int* in_sizes, int n_in,
                              void* d_out, int out_size, void* d_ws, size_t ws_size,
                              hipStream_t stream) {
  (void)in_sizes; (void)n_in; (void)out_size; (void)ws_size;
  const float* node_emb   = (const float*)d_in[0];
  const float* edge_emb   = (const float*)d_in[1];
  const float* lin_w      = (const float*)d_in[2];
  const float* lin_b      = (const float*)d_in[3];
  const float* alpha_w    = (const float*)d_in[4];
  const float* alpha_b    = (const float*)d_in[5];
  const float* beta_w     = (const float*)d_in[6];
  const float* beta_b     = (const float*)d_in[7];
  const float* wr_w       = (const float*)d_in[8];
  const float* wr_b       = (const float*)d_in[9];
  const float* conv_w     = (const float*)d_in[10];
  const float* conv_b     = (const float*)d_in[11];
  const float* mlp_w      = (const float*)d_in[12];
  const float* mlp_b      = (const float*)d_in[13];
  const int*   visit_node = (const int*)d_in[14];
  const int*   ehr        = (const int*)d_in[15];
  const int*   nid        = (const int*)d_in[16];
  const int*   eid        = (const int*)d_in[17];
  const int*   src        = (const int*)d_in[18];
  const int*   dst        = src + ETOT;
  const int*   bat        = (const int*)d_in[19];

  char* ws = (char*)d_ws;
  size_t off = 0;
  auto alloc = [&](size_t bytes) -> void* {
    void* p = ws + off;
    off = (off + bytes + 255) & ~(size_t)255;
    return p;
  };
  float*  x        = (float*)alloc((size_t)NTOT * HH * 4);
  float*  agg      = (float*)alloc((size_t)NTOT * HH * 4);
  float*  xproj    = (float*)alloc((size_t)NVV * HH * 4);
  float*  logits   = (float*)alloc((size_t)BN * VVIS * NVP * 4);
  float*  attn     = (float*)alloc((size_t)LLY * BN * NVV * 4);
  float*  beta     = (float*)alloc((size_t)LLY * BN * VVIS * 4);
  float*  wrel     = (float*)alloc((size_t)LLY * NEE * 4);
  float*  u        = (float*)alloc((size_t)LLY * HH * 4);
  float*  cvec     = (float*)alloc((size_t)LLY * 4);
  float*  xgraph   = (float*)alloc((size_t)BN * HH * 4);
  float*  xne      = (float*)alloc((size_t)BN * HH * 4);
  float*  xnode    = (float*)alloc((size_t)BN * HH * 4);
  bf16_t* nodeembb = (bf16_t*)alloc((size_t)NVP * HH * 2);          // padded rows
  bf16_t* linb     = (bf16_t*)alloc((size_t)HH * HH * 2);
  bf16_t* convb    = (bf16_t*)alloc((size_t)LLY * HH * HH * 2);
  bf16_t* alphab   = (bf16_t*)alloc((size_t)LLY * NVP * NVP * 2);   // padded rows+cols
  bf16_t* vnb      = (bf16_t*)alloc((size_t)BN * VVIS * NVP * 2);   // padded cols

  // --- precompute tables (zero-padded so WMMA loaders are branch-free) ---
  cvt_node_kernel<<<(int)(((long)NVP * HH + 255) / 256), 256, 0, stream>>>(node_emb, nodeembb);
  cvt_bf16_kernel<<<(HH * HH + 255) / 256, 256, 0, stream>>>(lin_w, linb, (long)HH * HH);
  cvt_bf16_kernel<<<(LLY * HH * HH + 255) / 256, 256, 0, stream>>>(conv_w, convb, (long)LLY * HH * HH);
  cvt_vn_kernel<<<(int)(((long)BN * VVIS * NVP + 255) / 256), 256, 0, stream>>>(visit_node, vnb);
  cvt_alpha_kernel<<<(int)(((long)LLY * NVP * NVP + 255) / 256), 256, 0, stream>>>(alpha_w, alphab);
  u_kernel<<<LLY, HH, 0, stream>>>(lin_w, lin_b, wr_w, wr_b, u, cvec);
  wrel_kernel<<<(LLY * NEE + 255) / 256, 256, 0, stream>>>(edge_emb, u, cvec, wrel);
  beta_kernel<<<(LLY * BN * VVIS + 127) / 128, 128, 0, stream>>>(visit_node, beta_w, beta_b, beta);

  // xproj[NVV][HH] = node_emb @ lin_w^T + lin_b   (WMMA)
  gemm_abt_kernel<<<dim3((NVP / 16 + 3) / 4, HH / 16), 128, 0, stream>>>(
      nodeembb, HH, NVV, linb, HH, HH, HH, lin_b, xproj, HH);

  // attention per layer (WMMA logits GEMM + softmax-reduce)
  for (int l = 0; l < LLY; ++l) {
    gemm_abt_kernel<<<dim3((BN * VVIS / 16 + 3) / 4, NVP / 16), 128, 0, stream>>>(
        vnb, NVP, BN * VVIS, alphab + (size_t)l * NVP * NVP, NVP, NVV, NVP,
        alpha_b + l * NVV, logits, NVP);
    attn_kernel<<<(BN * NVV + 255) / 256, 256, 0, stream>>>(
        logits, beta + l * BN * VVIS, attn + (size_t)l * BN * NVV);
  }
  // x init: gather projected rows
  xinit_kernel<<<(int)(((long)NTOT * 32 + 255) / 256), 256, 0, stream>>>(
      (const float4*)xproj, nid, (float4*)x);

  // GNN layers: message scatter + conv GEMM (WMMA, in-place)
  for (int l = 0; l < LLY; ++l) {
    hipMemsetAsync(agg, 0, (size_t)NTOT * HH * 4, stream);
    message_kernel<<<4096, 128, 0, stream>>>(x, agg, src, dst, nid, eid, bat,
                                             attn + (size_t)l * BN * NVV, wrel + l * NEE);
    conv_kernel<<<(NTOT / 16 + 7) / 8, 256, 0, stream>>>(x, agg,
                                                         convb + (size_t)l * HH * HH,
                                                         conv_b + l * HH);
  }

  // pooling + node branch + head
  hipMemsetAsync(xgraph, 0, (size_t)BN * HH * 4, stream);
  pool_kernel<<<dim3(BN, 8), HH, 0, stream>>>(x, xgraph);
  xnode_emb_kernel<<<BN, HH, 0, stream>>>(ehr, node_emb, xne);
  xnode_proj_kernel<<<(BN * HH + 255) / 256, 256, 0, stream>>>(xne, lin_w, lin_b, xnode);
  final_kernel<<<(BN * OUTD + 127) / 128, 128, 0, stream>>>(xgraph, xnode, mlp_w, mlp_b,
                                                            (float*)d_out);
}
